// BiMamba2Dv3_31971736552185
// MI455X (gfx1250) — compile-verified, matrix operand order checked by reading, not run on hardware
//
#include <hip/hip_runtime.h>
#include <hip/hip_bf16.h>

typedef __attribute__((ext_vector_type(16))) __bf16 v16bf;
typedef __attribute__((ext_vector_type(8)))  float  v8f;

#define B_DIM 2
#define H_DIM 64
#define W_DIM 64
#define L_DIM 4096
#define DM    256
#define DI    512
#define NST   16
#define RNK   16
#define MROWS (B_DIM * L_DIM)   // 8192
#define NBLK  4                 // 16x64 output strip per wave (4 WMMA tiles)
#define KT    32                // K-step (one WMMA depth)

__device__ __forceinline__ float sigm_(float x) { return 1.0f / (1.0f + __expf(-x)); }
__device__ __forceinline__ float silu_(float x) { return x * sigm_(x); }
__device__ __forceinline__ float softplus_(float x) { return (x > 20.0f) ? x : log1pf(__expf(x)); }

// Build a bf16 A/B WMMA fragment (16x32 / 32x16, 16-bit layout from ISA 7.12.2):
// lanes 0-15 hold K {kbase+0..7, kbase+16..23}; lanes 16-31 hold K {kbase+8..15, kbase+24..31}.
template <typename F>
__device__ __forceinline__ v16bf make_frag(int lane, int kbase, F f) {
    const int hi = (lane >> 4) * 8;
    v16bf v;
#pragma unroll
    for (int i = 0; i < 8; ++i) {
        v[i]     = (__bf16)f(kbase + hi + i);
        v[i + 8] = (__bf16)f(kbase + 16 + hi + i);
    }
    return v;
}

// Async copy of one f32 element into LDS (CDNA5 GLOBAL_LOAD_ASYNC_TO_LDS_B32, ASYNCcnt).
__device__ __forceinline__ void async_ld_b32(uint32_t lds_off, const float* gp) {
    asm volatile("global_load_async_to_lds_b32 %0, %1, off"
                 :: "v"(lds_off), "v"((uint64_t)(uintptr_t)gp)
                 : "memory");
}
__device__ __forceinline__ void wait_async0() {
    asm volatile("s_wait_asynccnt 0" ::: "memory");
}

// ---------------- Kernel 1: xz = x @ in_proj_w ; split into xi / z ----------------
// Block = 8 waves x (16 rows) = 128 rows, all sharing one 64-wide column group.
// B chunks (32x64 f32) staged to LDS with async copies, double-buffered.
__global__ void k_inproj(const float* __restrict__ x, const float* __restrict__ w,
                         float* __restrict__ xi, float* __restrict__ z) {
    __shared__ float bstage[2][KT * 64];     // 2 x 8KB
    const int tidb = threadIdx.x;            // 0..255
    const int lane = tidb & 31;
    const int ln15 = lane & 15;
    const int wav  = tidb >> 5;              // 0..7
    const int NG   = (2 * DI) / 64;          // 16 column groups
    const int tm   = ((blockIdx.x / NG) * 8 + wav) * 16;
    const int tn   = (blockIdx.x % NG) * 64;
    const int m    = tm + ln15;

    auto stage = [&](int buf, int k0) {
        const uint32_t lbase = (uint32_t)(uintptr_t)&bstage[buf][0];
#pragma unroll
        for (int i = 0; i < 8; ++i) {
            const int idx = i * 256 + tidb;  // 0..2047 over 32x64 chunk
            const int kk  = idx >> 6;
            const int c   = idx & 63;
            async_ld_b32(lbase + (uint32_t)idx * 4u,
                         &w[(size_t)(k0 + kk) * (2 * DI) + tn + c]);
        }
    };

    v8f acc[NBLK] = {};
    stage(0, 0);
    int buf = 0;
    for (int k0 = 0; k0 < DM; k0 += KT, buf ^= 1) {
        wait_async0();
        __syncthreads();
        if (k0 + KT < DM) stage(buf ^ 1, k0 + KT);
        if (k0 + KT < DM) __builtin_prefetch(&x[(size_t)m * DM + k0 + KT], 0, 1);
        v16bf a = make_frag(lane, k0, [&](int k) { return x[(size_t)m * DM + k]; });
        const float* bs = &bstage[buf][0];
#pragma unroll
        for (int j = 0; j < NBLK; ++j) {
            v16bf b = make_frag(lane, 0, [&](int k) { return bs[k * 64 + j * 16 + ln15]; });
            acc[j] = __builtin_amdgcn_wmma_f32_16x16x32_bf16(false, a, false, b, (short)0,
                                                             acc[j], false, false);
        }
    }
    const int hi = (lane >> 4) * 8;
#pragma unroll
    for (int j = 0; j < NBLK; ++j) {
        const int n = tn + j * 16 + ln15;
#pragma unroll
        for (int v = 0; v < 8; ++v) {
            const int row = tm + hi + v;
            if (n < DI) xi[(size_t)row * DI + n]       = acc[j][v];
            else        z[(size_t)row * DI + (n - DI)] = acc[j][v];
        }
    }
}

// --------- Kernel 2: 3x3 conv 512->512 as implicit GEMM (K = 9*512), + bias + SiLU ---------
// Same block structure; conv_w chunks staged to LDS asynchronously (stride-9 gather done once
// per 128 output rows instead of per 16).
__global__ void k_conv(const float* __restrict__ xi, const float* __restrict__ cw,
                       const float* __restrict__ cb, float* __restrict__ xc) {
    __shared__ float bstage[2][KT * 64];
    const int tidb = threadIdx.x;
    const int lane = tidb & 31;
    const int ln15 = lane & 15;
    const int wav  = tidb >> 5;
    const int NG   = DI / 64;                // 8 column groups
    const int tm   = ((blockIdx.x / NG) * 8 + wav) * 16;
    const int tn   = (blockIdx.x % NG) * 64;
    const int m    = tm + ln15;
    const int b    = m >> 12;
    const int l    = m & (L_DIM - 1);
    const int h    = l >> 6;
    const int ww0  = l & 63;

    auto stage = [&](int buf, int k0) {      // chunk has fixed tap t = k0>>9
        const uint32_t lbase = (uint32_t)(uintptr_t)&bstage[buf][0];
        const int i0 = k0 & (DI - 1);
        const int t  = k0 >> 9;
#pragma unroll
        for (int i = 0; i < 8; ++i) {
            const int idx = i * 256 + tidb;
            const int kk  = idx >> 6;
            const int c   = idx & 63;
            async_ld_b32(lbase + (uint32_t)idx * 4u,
                         &cw[(size_t)((tn + c) * DI + i0 + kk) * 9 + t]);
        }
    };

    v8f acc[NBLK] = {};
    stage(0, 0);
    int buf = 0;
    for (int k0 = 0; k0 < 9 * DI; k0 += KT, buf ^= 1) {
        wait_async0();
        __syncthreads();
        if (k0 + KT < 9 * DI) stage(buf ^ 1, k0 + KT);
        v16bf a = make_frag(lane, k0, [&](int k) {
            const int i  = k & (DI - 1);
            const int t  = k >> 9;           // 0..8
            const int hh = h + t / 3 - 1;
            const int ww = ww0 + t % 3 - 1;
            if ((unsigned)hh >= (unsigned)H_DIM || (unsigned)ww >= (unsigned)W_DIM) return 0.0f;
            return xi[(size_t)((b << 12) + (hh << 6) + ww) * DI + i];
        });
        const float* bs = &bstage[buf][0];
#pragma unroll
        for (int j = 0; j < NBLK; ++j) {
            v16bf bb = make_frag(lane, 0, [&](int k) { return bs[k * 64 + j * 16 + ln15]; });
            acc[j] = __builtin_amdgcn_wmma_f32_16x16x32_bf16(false, a, false, bb, (short)0,
                                                             acc[j], false, false);
        }
    }
    const int hi = (lane >> 4) * 8;
#pragma unroll
    for (int j = 0; j < NBLK; ++j) {
        const int   n    = tn + j * 16 + ln15;
        const float bias = cb[n];
#pragma unroll
        for (int v = 0; v < 8; ++v) {
            const int row = tm + hi + v;
            xc[(size_t)row * DI + n] = silu_(acc[j][v] + bias);
        }
    }
}

// ---------------- Kernel 3: x_dbl = xc @ xproj_w^T (8192x512 @ 512x192) ----------------
__global__ void k_xproj(const float* __restrict__ xc, const float* __restrict__ pw,
                        float* __restrict__ xdbl) {
    const int lane = threadIdx.x & 31;
    const int ln15 = lane & 15;
    const int wid  = blockIdx.x * (blockDim.x >> 5) + (threadIdx.x >> 5);
    const int NT   = 192 / 16;               // 12 column tiles
    const int tm   = (wid / NT) * 16;
    const int tn   = (wid % NT) * 16;
    const int m    = tm + ln15;
    const int n    = tn + ln15;
    v8f acc = {};
    for (int k0 = 0; k0 < DI; k0 += KT) {
        if (k0 + KT < DI) __builtin_prefetch(&xc[(size_t)m * DI + k0 + KT], 0, 1);
        v16bf a = make_frag(lane, k0, [&](int k) { return xc[(size_t)m * DI + k]; });
        v16bf b = make_frag(lane, k0, [&](int k) { return pw[(size_t)n * DI + k]; });
        acc = __builtin_amdgcn_wmma_f32_16x16x32_bf16(false, a, false, b, (short)0, acc, false, false);
    }
    const int hi = (lane >> 4) * 8;
#pragma unroll
    for (int v = 0; v < 8; ++v) xdbl[(size_t)(tm + hi + v) * 192 + n] = acc[v];
}

// ---------------- Kernel 4: zero the y accumulator ----------------
__global__ void k_zero(float* __restrict__ p, int nelem) {
    for (int i = blockIdx.x * blockDim.x + threadIdx.x; i < nelem; i += gridDim.x * blockDim.x)
        p[i] = 0.0f;
}

// ------- Kernel 5: 4-direction selective scan; 16 lanes per (dir,b,d), lane = state n -------
__global__ void k_scan(const float* __restrict__ xc, const float* __restrict__ xdbl,
                       const float* __restrict__ dt_w, const float* __restrict__ dt_b,
                       const float* __restrict__ A_logs, const float* __restrict__ Ds,
                       float* __restrict__ y) {
    const int tid  = blockIdx.x * blockDim.x + threadIdx.x;
    const int g    = tid >> 4;               // group id: 0..4095
    const int n    = tid & 15;               // state index
    const int kdir = g >> 10;
    const int b    = (g >> 9) & 1;
    const int d    = g & (DI - 1);
    const int kd   = kdir * DI + d;

    const float a_dn  = -__expf(A_logs[(size_t)kd * NST + n]);
    const float dtw_n = dt_w[(size_t)kd * RNK + n];
    const float dtb   = dt_b[kd];
    const float Dval  = Ds[kd];
    const int   uch   = (kdir & 1) ? (DI - 1 - d) : d;   // xs[1],xs[3] are channel-reversed

    int roff, boff, coff;
    switch (kdir) {
        case 0:  roff = 0;   boff = 32;  coff = 64;  break;
        case 1:  roff = 16;  boff = 48;  coff = 80;  break;
        case 2:  roff = 96;  boff = 128; coff = 160; break;
        default: roff = 112; boff = 144; coff = 176; break;
    }

    float hstate = 0.0f;
    for (int l = 0; l < L_DIM; ++l) {
        // wh directions traverse W-major: scan index l = w*64+h -> spatial (l&63)*64 + (l>>6)
        const int pos = (kdir < 2) ? l : (((l & 63) << 6) | (l >> 6));
        const float* xr = xdbl + (size_t)(b * L_DIM + pos) * 192;
        const float uval = xc[(size_t)(b * L_DIM + pos) * DI + uch];

        float p = xr[roff + n] * dtw_n;      // r == n == 16: lane-parallel dot
#pragma unroll
        for (int s = 8; s >= 1; s >>= 1) p += __shfl_xor(p, s, 16);
        const float delta = softplus_(p + dtb);

        const float bt = xr[boff + n];
        const float ct = xr[coff + n];
        hstate = __expf(delta * a_dn) * hstate + (delta * uval) * bt;

        float yv = hstate * ct;
#pragma unroll
        for (int s = 8; s >= 1; s >>= 1) yv += __shfl_xor(yv, s, 16);

        if (n == 0)
            atomicAdd(&y[(size_t)(b * L_DIM + l) * DI + d], yv + Dval * uval);
    }
}

// ------- Kernel 6: out = (y * silu(z)) @ out_proj_w ; gating fused into A loads -------
__global__ void k_outproj(const float* __restrict__ y, const float* __restrict__ z,
                          const float* __restrict__ ow, float* __restrict__ out) {
    const int lane = threadIdx.x & 31;
    const int ln15 = lane & 15;
    const int wid  = blockIdx.x * (blockDim.x >> 5) + (threadIdx.x >> 5);
    const int NG   = DM / (16 * NBLK);       // 4 column groups of 64
    const int tm   = (wid / NG) * 16;
    const int tn   = (wid % NG) * (16 * NBLK);
    const int m    = tm + ln15;
    v8f acc[NBLK] = {};
    for (int k0 = 0; k0 < DI; k0 += KT) {
        if (k0 + KT < DI) {
            __builtin_prefetch(&y[(size_t)m * DI + k0 + KT], 0, 1);
            __builtin_prefetch(&z[(size_t)m * DI + k0 + KT], 0, 1);
        }
        v16bf a = make_frag(lane, k0, [&](int k) {
            return y[(size_t)m * DI + k] * silu_(z[(size_t)m * DI + k]);
        });
#pragma unroll
        for (int j = 0; j < NBLK; ++j) {
            const int n = tn + j * 16 + ln15;
            v16bf b = make_frag(lane, k0, [&](int k) { return ow[(size_t)k * DM + n]; });
            acc[j] = __builtin_amdgcn_wmma_f32_16x16x32_bf16(false, a, false, b, (short)0,
                                                             acc[j], false, false);
        }
    }
    const int hi = (lane >> 4) * 8;
#pragma unroll
    for (int j = 0; j < NBLK; ++j) {
        const int n = tn + j * 16 + ln15;
#pragma unroll
        for (int v = 0; v < 8; ++v) out[(size_t)(tm + hi + v) * DM + n] = acc[j][v];
    }
}

extern "C" void kernel_launch(void* const* d_in, const int* in_sizes, int n_in,
                              void* d_out, int out_size, void* d_ws, size_t ws_size,
                              hipStream_t stream) {
    (void)in_sizes; (void)n_in; (void)out_size; (void)ws_size;
    const float* x         = (const float*)d_in[0];
    const float* in_proj_w = (const float*)d_in[1];
    const float* conv_w    = (const float*)d_in[2];
    const float* conv_b    = (const float*)d_in[3];
    const float* xproj_w   = (const float*)d_in[4];
    const float* dt_w      = (const float*)d_in[5];
    const float* dt_b      = (const float*)d_in[6];
    const float* A_logs    = (const float*)d_in[7];
    const float* Ds        = (const float*)d_in[8];
    const float* out_proj  = (const float*)d_in[9];
    float* out = (float*)d_out;

    // workspace layout (floats)
    float* ws     = (float*)d_ws;
    float* buf_xi = ws;                                   // 8192*512
    float* buf_z  = buf_xi + (size_t)MROWS * DI;          // 8192*512
    float* buf_xc = buf_z  + (size_t)MROWS * DI;          // 8192*512
    float* buf_xd = buf_xc + (size_t)MROWS * DI;          // 8192*192
    float* buf_y  = buf_xd + (size_t)MROWS * 192;         // 8192*512

    const int threads = 256;               // 8 waves per block
    // 1) in_proj: 64 row groups x 16 column groups
    k_inproj<<<64 * 16, threads, 0, stream>>>(x, in_proj_w, buf_xi, buf_z);
    // 2) conv: 64 row groups x 8 column groups
    k_conv<<<64 * 8, threads, 0, stream>>>(buf_xi, conv_w, conv_b, buf_xc);
    // 3) x_dbl projection: 512 x 12 tiles
    k_xproj<<<(512 * 12) / 8, threads, 0, stream>>>(buf_xc, xproj_w, buf_xd);
    // 4) zero y accumulator
    k_zero<<<1024, threads, 0, stream>>>(buf_y, MROWS * DI);
    // 5) selective scan: 4096 groups * 16 lanes = 65536 threads
    k_scan<<<(4096 * 16) / threads, threads, 0, stream>>>(buf_xc, buf_xd, dt_w, dt_b,
                                                          A_logs, Ds, buf_y);
    // 6) gated out_proj: 512 row tiles x 4 column groups
    k_outproj<<<(512 * 4) / 8, threads, 0, stream>>>(buf_y, buf_z, out_proj, out);
}
